// GCN_TCN_53154515256268
// MI455X (gfx1250) — compile-verified, matrix operand order checked by reading, not run on hardware
//
#include <hip/hip_runtime.h>
#include <hip/hip_bf16.h>

// ---------------- types for WMMA fragments ----------------
typedef _Float16 half8  __attribute__((ext_vector_type(8)));
typedef _Float16 half16 __attribute__((ext_vector_type(16)));
typedef float    v8f    __attribute__((ext_vector_type(8)));
typedef unsigned int u32x4 __attribute__((ext_vector_type(4)));
typedef int          i32x8 __attribute__((ext_vector_type(8)));
typedef int          i32x4 __attribute__((ext_vector_type(4)));

union HalfU {
    half16   v;
    half8    h[2];
    _Float16 e[16];
};
union Half8U {
    half8    v;
    _Float16 e[8];
};

#define Tn   2048
#define Nn   24
#define Bn   16
#define BNn  (Bn * Nn)          // 384
#define C64  64
#define TOTE ((size_t)BNn * C64 * Tn)   // 50,331,648 elements per activation tensor

#if __has_builtin(__builtin_amdgcn_tensor_load_to_lds) && __has_builtin(__builtin_amdgcn_s_wait_tensorcnt)
#define USE_TDM 1
#else
#define USE_TDM 0
#endif

// =====================================================================
// Kernel 1: prep — normalized Laplacian, Chebyshev stack, fused GC weight
//           W2[(n*64+o)][m] (f16, m padded to 32), and conv-weight repack
//           Wc[o][j*64+c] (f16, K = 192) for the 4 temporal convs.
// =====================================================================
__global__ void k_prep(const float* __restrict__ adj, const float* __restrict__ theta,
                       const float* __restrict__ w1a, const float* __restrict__ w1b,
                       const float* __restrict__ w2a, const float* __restrict__ w2b,
                       _Float16* __restrict__ W2, _Float16* __restrict__ Wc)
{
    __shared__ float Ls[24][24], T2s[24][24], dis[24];
    int tid = threadIdx.x;

    if (tid < 24) {
        float s = 0.f;
        for (int j = 0; j < 24; j++) s += adj[tid * 24 + j];
        dis[tid] = (s > 0.f) ? rsqrtf(s) : 0.f;
    }
    __syncthreads();
    // L = dis[:,None] * adj^T * dis[None,:]
    for (int i = tid; i < 576; i += blockDim.x) {
        int r = i / 24, c = i % 24;
        Ls[r][c] = dis[r] * adj[c * 24 + r] * dis[c];
    }
    __syncthreads();
    // T2 = 2*L@L - I
    for (int i = tid; i < 576; i += blockDim.x) {
        int r = i / 24, c = i % 24;
        float s = 0.f;
        for (int m = 0; m < 24; m++) s += Ls[r][m] * Ls[m][c];
        T2s[r][c] = 2.f * s - (r == c ? 1.f : 0.f);
    }
    __syncthreads();
    // W2[(n*64+o)*32 + m]
    for (int i = tid; i < 1536 * 32; i += blockDim.x) {
        int row = i >> 5, m = i & 31;
        int n = row >> 6, o = row & 63;
        float v = 0.f;
        if (m < 24) {
            v = (m == n ? theta[o] : 0.f) + Ls[m][n] * theta[64 + o] + T2s[m][n] * theta[128 + o];
        }
        W2[i] = (_Float16)v;
    }
    // Conv weight repack: Wc[s][o*192 + j*64 + c] = w_s[o,c,0,j]
    const float* ws[4] = { w1a, w1b, w2a, w2b };
    for (int s = 0; s < 4; s++) {
        for (int i = tid; i < 64 * 192; i += blockDim.x) {
            int o = i / 192, rem = i % 192;
            int j = rem >> 6, c = rem & 63;
            Wc[s * 64 * 192 + i] = (_Float16)ws[s][o * 192 + c * 3 + j];
        }
    }
}

// =====================================================================
// Kernel 2: graph conv as WMMA GEMM. Per batch b:
//   gc(1536 x T) = W2(1536 x 24pad32) @ X(24 x T),  X[m,t] = batch_x[b,t,m]
// Output f16 in ACTIVATION layout [bn][t][c]: one contiguous b128 store
// per lane (8 consecutive channels).
// =====================================================================
__global__ void k_gc(const float* __restrict__ bx, const _Float16* __restrict__ W2,
                     _Float16* __restrict__ gc)
{
    int b    = blockIdx.z;
    int mt   = blockIdx.y;                 // 0..95  (row tile of 1536)
    int wave = threadIdx.x >> 5;
    int lane = threadIdx.x & 31;
    int hi = lane >> 4, ln = lane & 15;
    int t0 = blockIdx.x * 128 + wave * 16;

    // A fragment: rows of W2 (16x32 f16)
    HalfU a;
    const _Float16* wrow = W2 + (size_t)(mt * 16 + ln) * 32;
    a.h[0] = *(const half8*)(wrow + hi * 8);
    a.h[1] = *(const half8*)(wrow + 16 + hi * 8);

    // B fragment: column t, K = m (pad m>=24 with 0)
    HalfU bf;
    int t = t0 + ln;
    const float* xcol = bx + ((size_t)b * Tn + t) * Nn;
#pragma unroll
    for (int q = 0; q < 8; q++) {
        int m0 = hi * 8 + q;
        bf.e[q]     = (_Float16)(m0 < 24 ? xcol[m0] : 0.f);
        int m1 = 16 + hi * 8 + q;
        bf.e[8 + q] = (_Float16)(m1 < 24 ? xcol[m1] : 0.f);
    }

    v8f acc = {};
    acc = __builtin_amdgcn_wmma_f32_16x16x32_f16(false, a.v, false, bf.v,
                                                 (short)0, acc, false, false);

    // D tile: M rows = channels n*64+o ; this lane holds 8 consecutive o
    int n_    = mt >> 2;                       // node index (16 rows stay in one n)
    int obase = (mt & 3) * 16 + hi * 8;        // first channel of this lane's 8
    Half8U d;
#pragma unroll
    for (int r = 0; r < 8; r++) d.e[r] = (_Float16)acc[r];
    *(half8*)(gc + ((size_t)(b * Nn + n_) * Tn + (t0 + ln)) * C64 + obase) = d.v;
}

// =====================================================================
// Kernel 3: causal dilated temporal conv as WMMA GEMM + bias + f32 stats.
//   Y(64 x 128tile) = Wc(64 x 192) @ im2col(192 x 128), K-order j*64+c
// Activations are [bn][t][c] (128B rows) -> the input tile is one
// contiguous global region; staged into LDS by the Tensor Data Mover with
// hardware padding (row stride 144B = bank-conflict-free b128 frag reads).
// =====================================================================
template <int DIL>
__global__ void k_conv(const _Float16* __restrict__ X, const _Float16* __restrict__ Wc,
                       const float* __restrict__ bias, _Float16* __restrict__ Y,
                       float* __restrict__ stats)
{
    constexpr int TT = 128, HALO = 2 * DIL, ROW = 72;   // ROW in halves (144 B)
    __shared__ __align__(16) _Float16 lds[(TT + 4) * ROW];

    int bn = blockIdx.y;
    int t0 = blockIdx.x * TT;
    int tid = threadIdx.x;
    const _Float16* Xb = X + (size_t)bn * Tn * C64;

#if USE_TDM
    if (t0 == 0) {  // zero the causal halo rows; TDM fills the rest
        for (int i = tid; i < HALO * ROW; i += 256) lds[i] = (_Float16)0.f;
    }
    if (tid < 32) {   // wave 0 issues the DMA (TDM ignores EXEC; one issue/wave)
        int rows      = (t0 == 0) ? TT : (TT + HALO);
        long gofs     = (t0 == 0) ? 0 : (long)(t0 - HALO) * C64;
        int ldsRowOff = (t0 == 0) ? HALO : 0;
        unsigned long long ga = (unsigned long long)(uintptr_t)(Xb + gofs);
        unsigned int       la = (unsigned int)(uintptr_t)(&lds[ldsRowOff * ROW]);
        u32x4 g0;
        g0[0] = 1u;                                        // count=1, user mode
        g0[1] = la;                                        // lds_addr
        g0[2] = (unsigned int)ga;                          // global_addr[31:0]
        g0[3] = (unsigned int)(ga >> 32) | (2u << 30);     // global_addr[56:32] | type=2
        i32x8 g1;
        g1[0] = (1 << 16) | (1 << 20) | (4 << 22) | (3 << 25); // 2B elems, pad_en, 32dw interval, 4dw pad
        g1[1] = (int)(64u << 16);                          // tensor_dim0 = 64
        g1[2] = (int)(((unsigned)rows & 0xFFFFu) << 16);   // tensor_dim1 = rows (lo16)
        g1[3] = (int)(64u << 16);                          // dim1 hi16=0 | tile_dim0 = 64
        g1[4] = rows;                                      // tile_dim1 = rows, tile_dim2 = 0
        g1[5] = 64;                                        // tensor_dim0_stride = 64
        g1[6] = 0;
        g1[7] = 0;
        i32x4 gz4 = {0, 0, 0, 0};
        i32x8 gz8 = {0, 0, 0, 0, 0, 0, 0, 0};
        __builtin_amdgcn_tensor_load_to_lds(g0, g1, gz4, gz4, gz8, 0);
        __builtin_amdgcn_s_wait_tensorcnt(0);
    }
    __syncthreads();
#else
    for (int i = tid; i < (TT + HALO) * 64; i += 256) {
        int tl = i >> 6, c = i & 63;
        int tg = t0 - HALO + tl;
        lds[tl * ROW + c] = (tg >= 0) ? Xb[(size_t)tg * C64 + c] : (_Float16)0.f;
    }
    __syncthreads();
#endif

    int wave = tid >> 5, lane = tid & 31;
    int mt = wave & 3, nh = wave >> 2;
    int hi = lane >> 4, ln = lane & 15;

    // A fragments (weights), loaded once: 6 K-steps
    HalfU a[6];
    {
        const _Float16* wr = Wc + (size_t)(mt * 16 + ln) * 192;
#pragma unroll
        for (int ks = 0; ks < 6; ks++) {
            a[ks].h[0] = *(const half8*)(wr + ks * 32 + hi * 8);
            a[ks].h[1] = *(const half8*)(wr + ks * 32 + 16 + hi * 8);
        }
    }
    int obase = mt * 16 + hi * 8;      // this lane's first output channel
    float bias_r[8];
#pragma unroll
    for (int r = 0; r < 8; r++) bias_r[r] = bias[obase + r];

    float sum1[8] = {0,0,0,0,0,0,0,0}, sum2[8] = {0,0,0,0,0,0,0,0};

#pragma unroll
    for (int nt = 0; nt < 4; nt++) {
        int tl_out = nh * 64 + nt * 16 + ln;      // local output t
        v8f acc = {};
#pragma unroll
        for (int ks = 0; ks < 6; ks++) {
            int j = ks >> 1;
            int cbase = (ks & 1) * 32;
            // causal tap: input t = out_t - (2-j)*DIL  ->  LDS row tl_out + j*DIL
            const _Float16* p = &lds[(tl_out + j * DIL) * ROW + cbase + hi * 8];
            HalfU bf;
            bf.h[0] = *(const half8*)p;
            bf.h[1] = *(const half8*)(p + 16);
            acc = __builtin_amdgcn_wmma_f32_16x16x32_f16(false, a[ks].v, false, bf.v,
                                                         (short)0, acc, false, false);
        }
        int tg = t0 + tl_out;
        Half8U d;
#pragma unroll
        for (int r = 0; r < 8; r++) {
            float y = acc[r] + bias_r[r];
            sum1[r] += y;
            sum2[r] += y * y;
            d.e[r] = (_Float16)y;
        }
        *(half8*)(Y + ((size_t)bn * Tn + tg) * C64 + obase) = d.v;   // one b128 store
    }

    // per-channel reduction across the 16 lanes of each half-wave, then atomics
#pragma unroll
    for (int r = 0; r < 8; r++) {
        float s1 = sum1[r], s2 = sum2[r];
#pragma unroll
        for (int m = 8; m >= 1; m >>= 1) {
            s1 += __shfl_xor(s1, m, 32);
            s2 += __shfl_xor(s2, m, 32);
        }
        if (ln == 0) {
            atomicAdd(&stats[obase + r], s1);
            atomicAdd(&stats[64 + obase + r], s2);
        }
    }
}

// =====================================================================
// Kernel 4: finalize BN stats -> scale/shift per channel
// =====================================================================
__global__ void k_statfin(const float* __restrict__ stats, const float* __restrict__ g,
                          const float* __restrict__ be, float* __restrict__ ss)
{
    int o = threadIdx.x;
    if (o >= 64) return;
    const float inv = 1.f / (float)((size_t)Bn * Nn * Tn);
    float mu  = stats[o] * inv;
    float var = stats[64 + o] * inv - mu * mu;
    float sc  = g[o] * rsqrtf(var + 1e-5f);
    ss[o]      = sc;
    ss[64 + o] = be[o] - mu * sc;
}

// =====================================================================
// Kernel 5: elementwise BN+ReLU (optional residual + outer ReLU for out0)
// Layout [bn][t][c]: 8 consecutive halves = 8 consecutive channels.
// =====================================================================
__global__ void k_elem(const _Float16* __restrict__ Yin, const float* __restrict__ ss,
                       const _Float16* __restrict__ Res, _Float16* __restrict__ Out)
{
    size_t base = ((size_t)blockIdx.x * 256 + threadIdx.x) * 8;
    int c0 = (int)(base & 63);
    Half8U in, res, ov;
    in.v = *(const half8*)(Yin + base);
    if (Res) res.v = *(const half8*)(Res + base);
#pragma unroll
    for (int q = 0; q < 8; q++) {
        float y = (float)in.e[q] * ss[c0 + q] + ss[64 + c0 + q];
        y = fmaxf(y, 0.f);
        if (Res) y = fmaxf(y + (float)res.e[q], 0.f);
        ov.e[q] = (_Float16)y;
    }
    *(half8*)(Out + base) = ov.v;
}

// =====================================================================
// Kernel 6: head — only last 25 time steps matter.
// =====================================================================
__global__ void k_final(const _Float16* __restrict__ Y2b, const float* __restrict__ ss,
                        const _Float16* __restrict__ Out0,
                        const float* __restrict__ wend, const float* __restrict__ bend,
                        const float* __restrict__ wend1, const float* __restrict__ bend1,
                        float* __restrict__ out)
{
    __shared__ float so[25][64];
    __shared__ float e1s[25][32];
    int bn = blockIdx.x;
    int b = bn / Nn, n = bn % Nn;
    int tid = threadIdx.x;            // 64 threads
    const int TB = Tn - 25;

    for (int i = tid; i < 25 * 64; i += 64) {
        int tt = i >> 6, c = i & 63;
        size_t idx = ((size_t)bn * Tn + TB + tt) * C64 + c;
        float y = fmaxf((float)Y2b[idx] * ss[c] + ss[64 + c], 0.f);
        so[tt][c] = y + (float)Out0[idx];
    }
    __syncthreads();
    for (int i = tid; i < 25 * 32; i += 64) {
        int tt = i >> 5, oc = i & 31;
        float s = bend[oc];
        for (int c = 0; c < 64; c++) s += wend[oc * 64 + c] * so[tt][c];
        e1s[tt][oc] = fmaxf(s, 0.f);
    }
    __syncthreads();
    if (tid < 24) {
        float s = bend1[0];
        for (int oc = 0; oc < 32; oc++)
            s += wend1[oc * 2] * e1s[tid][oc] + wend1[oc * 2 + 1] * e1s[tid + 1][oc];
        out[b * 576 + tid * 24 + n] = s;
    }
}

// =====================================================================
// Host-side orchestration
// =====================================================================
extern "C" void kernel_launch(void* const* d_in, const int* in_sizes, int n_in,
                              void* d_out, int out_size, void* d_ws, size_t ws_size,
                              hipStream_t stream)
{
    const float* batch_x = (const float*)d_in[0];
    const float* adj     = (const float*)d_in[1];
    const float* theta   = (const float*)d_in[2];
    const float* w1a = (const float*)d_in[3];  const float* b1a = (const float*)d_in[4];
    const float* g1a = (const float*)d_in[5];  const float* be1a= (const float*)d_in[6];
    const float* w1b = (const float*)d_in[7];  const float* b1b = (const float*)d_in[8];
    const float* g1b = (const float*)d_in[9];  const float* be1b= (const float*)d_in[10];
    const float* w2a = (const float*)d_in[11]; const float* b2a = (const float*)d_in[12];
    const float* g2a = (const float*)d_in[13]; const float* be2a= (const float*)d_in[14];
    const float* w2b = (const float*)d_in[15]; const float* b2b = (const float*)d_in[16];
    const float* g2b = (const float*)d_in[17]; const float* be2b= (const float*)d_in[18];
    const float* w_end  = (const float*)d_in[19]; const float* b_end  = (const float*)d_in[20];
    const float* w_end1 = (const float*)d_in[21]; const float* b_end1 = (const float*)d_in[22];
    float* outp = (float*)d_out;

    // ---- workspace layout (256B aligned regions) ----
    char* ws = (char*)d_ws;
    size_t off = 0;
    auto take = [&](size_t bytes) { char* p = ws + off; off = (off + bytes + 255) & ~(size_t)255; return p; };

    _Float16* W2    = (_Float16*)take(1536 * 32 * 2);        // GC fused weights
    _Float16* Wc    = (_Float16*)take(4 * 64 * 192 * 2);     // 4 repacked conv weights
    float*    stats = (float*)take(4 * 128 * 4);             // 4 stages x (sum,sumsq)[64]
    float*    ssb   = (float*)take(4 * 128 * 4);             // 4 stages x (scale,shift)[64]
    _Float16* gcb   = (_Float16*)take(TOTE * 2);
    _Float16* out0  = (_Float16*)take(TOTE * 2);
    _Float16* bufA  = (_Float16*)take(TOTE * 2);
    _Float16* bufB  = (_Float16*)take(TOTE * 2);
    if (off > ws_size) return;   // deterministic no-op if workspace too small

    // ---- stage 0: prep + zero stats ----
    k_prep<<<1, 256, 0, stream>>>(adj, theta, w1a, w1b, w2a, w2b, W2, Wc);
    (void)hipMemsetAsync(stats, 0, 4 * 128 * 4, stream);

    // ---- stage 1: graph conv (WMMA GEMM) -> gc f16 ----
    k_gc<<<dim3(Tn / 128, 96, Bn), 256, 0, stream>>>(batch_x, W2, gcb);

    const int EGRID = (int)(TOTE / (256 * 8));   // elementwise grid

    // ---- block 1, conv a (d=1) ----
    k_conv<1><<<dim3(Tn / 128, BNn), 256, 0, stream>>>(gcb, Wc + 0 * 64 * 192, b1a, bufA, stats + 0);
    k_statfin<<<1, 64, 0, stream>>>(stats + 0, g1a, be1a, ssb + 0);
    k_elem<<<EGRID, 256, 0, stream>>>(bufA, ssb + 0, nullptr, bufB);

    // ---- block 1, conv b (d=1) + residual with gc -> out0 ----
    k_conv<1><<<dim3(Tn / 128, BNn), 256, 0, stream>>>(bufB, Wc + 1 * 64 * 192, b1b, bufA, stats + 128);
    k_statfin<<<1, 64, 0, stream>>>(stats + 128, g1b, be1b, ssb + 128);
    k_elem<<<EGRID, 256, 0, stream>>>(bufA, ssb + 128, gcb, out0);

    // ---- block 2, conv a (d=2) ----
    k_conv<2><<<dim3(Tn / 128, BNn), 256, 0, stream>>>(out0, Wc + 2 * 64 * 192, b2a, bufA, stats + 256);
    k_statfin<<<1, 64, 0, stream>>>(stats + 256, g2a, be2a, ssb + 256);
    k_elem<<<EGRID, 256, 0, stream>>>(bufA, ssb + 256, nullptr, bufB);

    // ---- block 2, conv b (d=2) ----
    k_conv<2><<<dim3(Tn / 128, BNn), 256, 0, stream>>>(bufB, Wc + 3 * 64 * 192, b2b, bufA, stats + 384);
    k_statfin<<<1, 64, 0, stream>>>(stats + 384, g2b, be2b, ssb + 384);

    // ---- head: last 25 steps only ----
    k_final<<<BNn, 64, 0, stream>>>(bufA, ssb + 384, out0, w_end, b_end, w_end1, b_end1, outp);
}
